// RecurrentNet_88940182766040
// MI455X (gfx1250) — matrix operand order repaired and checked by reading, not compiled, not run on hardware
//
#include <hip/hip_runtime.h>
#include <hip/hip_bf16.h>

// ---------------------------------------------------------------------------
// Elman RNN on MI455X (gfx1250), bf16 WMMA with f32 accumulation.
//   Phase 0: convert weights to bf16 (L2-resident), combine biases, zero h0/ctr
//   Phase 1: X1 = relu(X @ W_in^T + b_in)            [32768 x 1024] bf16  (parallel)
//   Phase 2: U  = X1 @ W_ih^T + (b_ih+b_hh)          -> d_out hiddens f32 (parallel)
//   Phase 3: persistent kernel, 512 iterations with device-wide barrier:
//            h_t = relu(U_t + h_{t-1} @ W_hh^T), in place on d_out hiddens
//   Phase 4: O  = H @ W_out^T + b_out                -> d_out outputs f32 (parallel)
// ---------------------------------------------------------------------------

typedef __attribute__((ext_vector_type(16))) __bf16 v16bf;
typedef __attribute__((ext_vector_type(8)))  __bf16 v8bf;
typedef __attribute__((ext_vector_type(8)))  float  v8f;

#define T_STEPS 512
#define BATCH   64
#define NIN     256
#define NHID    1024
#define NOUT    256
#define SCAN_WGS 64   // persistent-scan workgroups (4 waves each -> 256 16x16 tiles)

// ---- WMMA fragment loaders (contiguous k order, k = 16*(lane/16) + elem, wave32) ----------

// A 16x32 bf16: lane L<16 -> M=L, K=k0+0..15 in elems 0..15
//               lane L>=16 -> M=L-16, K=k0+16..31.
__device__ __forceinline__ v16bf load_a_bf16(const __bf16* __restrict__ A, int lda, int lane) {
  const __bf16* p = A + (lane & 15) * lda + ((lane >> 4) << 4);
  v8bf lo = *(const v8bf*)p;          // 16B
  v8bf hi = *(const v8bf*)(p + 8);    // 16B
  return __builtin_shufflevector(lo, hi, 0,1,2,3,4,5,6,7,8,9,10,11,12,13,14,15);
}

__device__ __forceinline__ v16bf load_a_f32(const float* __restrict__ A, int lda, int lane) {
  const float* p = A + (lane & 15) * lda + ((lane >> 4) << 4);
  v16bf r;
#pragma unroll
  for (int i = 0; i < 8; ++i) { r[i] = (__bf16)p[i]; r[i + 8] = (__bf16)p[i + 8]; }
  return r;
}

// B 32x16 bf16: lane L<16 -> col N=L, K=k0+0..15; lane L>=16 -> col N=L-16, K=k0+16..31.
// W is [N,K] row-major, so B = W^T and each lane reads 16 contiguous bf16 (32B).
__device__ __forceinline__ v16bf load_b_bf16(const __bf16* __restrict__ Wnk, int ldw, int lane) {
  const __bf16* p = Wnk + (lane & 15) * ldw + ((lane >> 4) << 4);
  return *(const v16bf*)p;
}

// ---- Generic bf16-WMMA GEMM: C[M,N] = op(A[M,K] @ W[N,K]^T (+bias)) --------
// Software-pipelined K loop: fragments for k+32 are in flight while the WMMAs
// for k issue, so s_wait_loadcnt covers a full iteration of latency.
template<int K, int N, int NT, bool A_F32, bool RELU, bool HAS_BIAS,
         bool ST_F32, bool ST_BF16>
__global__ __launch_bounds__(128) void gemm_bf16_wmma(
    const void* __restrict__ A_, const __bf16* __restrict__ W,
    const float* __restrict__ bias, float* __restrict__ Cf, __bf16* __restrict__ Cb) {
  const int lane = threadIdx.x & 31;
  const int wave = threadIdx.x >> 5;
  const int m0   = blockIdx.y * 16;
  const int n0   = blockIdx.x * (NT * 64) + wave * (NT * 16);

  const int nlane = lane & 15;          // C/D: lane selects column
  const int mb    = (lane >> 4) << 3;   // C/D: upper half-lanes hold rows m0+8..15

  v8f acc[NT] = {};

  v16bf a_cur, a_nxt, b_cur[NT], b_nxt[NT];
  if (A_F32) a_cur = load_a_f32((const float*)A_ + (size_t)m0 * K, K, lane);
  else       a_cur = load_a_bf16((const __bf16*)A_ + (size_t)m0 * K, K, lane);
#pragma unroll
  for (int j = 0; j < NT; ++j)
    b_cur[j] = load_b_bf16(W + (size_t)(n0 + j * 16) * K, K, lane);
  a_nxt = a_cur;
#pragma unroll
  for (int j = 0; j < NT; ++j) b_nxt[j] = b_cur[j];

  for (int k0 = 0; k0 < K; k0 += 32) {
    const int k1 = k0 + 32;
    if (k1 < K) {
      if (A_F32) a_nxt = load_a_f32((const float*)A_ + (size_t)m0 * K + k1, K, lane);
      else       a_nxt = load_a_bf16((const __bf16*)A_ + (size_t)m0 * K + k1, K, lane);
#pragma unroll
      for (int j = 0; j < NT; ++j) {
        const __bf16* wnk = W + (size_t)(n0 + j * 16) * K + k1;
        b_nxt[j] = load_b_bf16(wnk, K, lane);
        __builtin_prefetch(wnk + 32, 0, 0);   // k+64 slab -> global_prefetch_b8
      }
    }
#pragma unroll
    for (int j = 0; j < NT; ++j)
      acc[j] = __builtin_amdgcn_wmma_f32_16x16x32_bf16(
          false, a_cur, false, b_cur[j], (short)0, acc[j], false, false);
    a_cur = a_nxt;
#pragma unroll
    for (int j = 0; j < NT; ++j) b_cur[j] = b_nxt[j];
  }

#pragma unroll
  for (int j = 0; j < NT; ++j) {
    const int col = n0 + j * 16 + nlane;
    const float bv = HAS_BIAS ? bias[col] : 0.0f;
#pragma unroll
    for (int r = 0; r < 8; ++r) {
      float v = acc[j][r] + bv;
      if (RELU) v = fmaxf(v, 0.0f);
      const size_t idx = (size_t)(m0 + mb + r) * N + col;
      if (ST_F32)  Cf[idx] = v;
      if (ST_BF16) Cb[idx] = (__bf16)v;
    }
  }
}

// ---- Phase 3: persistent scan kernel with device-wide barrier --------------
__device__ __forceinline__ void grid_barrier(unsigned* __restrict__ ctr, unsigned target) {
  __threadfence();        // flush this thread's h stores to device scope
  __syncthreads();        // all threads of WG have fenced
  if (threadIdx.x == 0) {
    unsigned v = __hip_atomic_fetch_add(ctr, 1u, __ATOMIC_ACQ_REL, __HIP_MEMORY_SCOPE_AGENT);
    if (v + 1u != target) {
      while (__hip_atomic_load(ctr, __ATOMIC_ACQUIRE, __HIP_MEMORY_SCOPE_AGENT) < target)
        __builtin_amdgcn_s_sleep(1);
    }
  }
  __syncthreads();        // release remaining waves of the WG
}

__global__ __launch_bounds__(128) void rnn_scan_persistent(
    const __bf16* __restrict__ Whh, float* __restrict__ hiddens,
    __bf16* __restrict__ Hbuf, unsigned* __restrict__ ctr) {
  const int lane  = threadIdx.x & 31;
  const int wave  = threadIdx.x >> 5;
  const int tile  = blockIdx.x * 4 + wave;       // 0..255 : full 64x1024 update
  const int m0    = (tile & 3) * 16;             // 4 row tiles (batch 64)
  const int n0    = (tile >> 2) * 16;            // 64 col tiles (hid 1024)
  const int nlane = lane & 15;
  const int mb    = (lane >> 4) << 3;

  for (int t = 0; t < T_STEPS; ++t) {
    const __bf16* Hp = Hbuf + (size_t)(t & 1) * BATCH * NHID;
    __bf16*       Hn = Hbuf + (size_t)((t + 1) & 1) * BATCH * NHID;
    float*        Ut = hiddens + (size_t)t * BATCH * NHID;

    // C init: U_t tile (in place in d_out)
    v8f acc;
    const float* cp = Ut + (size_t)(m0 + mb) * NHID + (n0 + nlane);
#pragma unroll
    for (int r = 0; r < 8; ++r) acc[r] = cp[(size_t)r * NHID];

    // pipelined K loop over h_{t-1} @ W_hh^T  (W_hh L2-resident across all t)
    v16bf a_cur = load_a_bf16(Hp + (size_t)m0 * NHID, NHID, lane);
    v16bf b_cur = load_b_bf16(Whh + (size_t)n0 * NHID, NHID, lane);
    v16bf a_nxt = a_cur, b_nxt = b_cur;
    for (int k0 = 0; k0 < NHID; k0 += 32) {
      const int k1 = k0 + 32;
      if (k1 < NHID) {
        a_nxt = load_a_bf16(Hp + (size_t)m0 * NHID + k1, NHID, lane);
        b_nxt = load_b_bf16(Whh + (size_t)n0 * NHID + k1, NHID, lane);
      }
      acc = __builtin_amdgcn_wmma_f32_16x16x32_bf16(
          false, a_cur, false, b_cur, (short)0, acc, false, false);
      a_cur = a_nxt;
      b_cur = b_nxt;
    }

    // epilogue: relu, write f32 h_t over U_t, bf16 copy for next iteration's A
#pragma unroll
    for (int r = 0; r < 8; ++r) {
      float v = fmaxf(acc[r], 0.0f);
      const size_t idx = (size_t)(m0 + mb + r) * NHID + (n0 + nlane);
      Ut[idx] = v;
      Hn[idx] = (__bf16)v;
    }

    grid_barrier(ctr, (unsigned)SCAN_WGS * (unsigned)(t + 1));
  }
}

// ---- Phase 0 helpers -------------------------------------------------------
__global__ void cvt_f32_bf16(const float* __restrict__ src, __bf16* __restrict__ dst, int n) {
  int i = blockIdx.x * blockDim.x + threadIdx.x;
  if (i < n) dst[i] = (__bf16)src[i];
}

__global__ void prep_bias_h0(const float* __restrict__ b_ih, const float* __restrict__ b_hh,
                             float* __restrict__ bias2, __bf16* __restrict__ h0,
                             unsigned* __restrict__ ctr) {
  int i = blockIdx.x * blockDim.x + threadIdx.x;
  if (i == 0) *ctr = 0u;                       // reset barrier counter every call
  if (i < NHID) bias2[i] = b_ih[i] + b_hh[i];
  if (i < BATCH * NHID) h0[i] = (__bf16)0.0f;
}

// ---------------------------------------------------------------------------
extern "C" void kernel_launch(void* const* d_in, const int* in_sizes, int n_in,
                              void* d_out, int out_size, void* d_ws, size_t ws_size,
                              hipStream_t stream) {
  (void)in_sizes; (void)n_in; (void)out_size; (void)ws_size;

  const float* X     = (const float*)d_in[0];   // [T,B,NIN]
  const float* W_in  = (const float*)d_in[2];   // [NHID,NIN]
  const float* b_in  = (const float*)d_in[3];
  const float* W_ih  = (const float*)d_in[4];   // [NHID,NHID]
  const float* b_ih  = (const float*)d_in[5];
  const float* W_hh  = (const float*)d_in[6];   // [NHID,NHID]
  const float* b_hh  = (const float*)d_in[7];
  const float* W_out = (const float*)d_in[8];   // [NOUT,NHID]
  const float* b_out = (const float*)d_in[9];

  float* hiddens = (float*)d_out;                              // [T*B, NHID]
  float* outputs = hiddens + (size_t)T_STEPS * BATCH * NHID;   // [T*B, NOUT]

  // Workspace layout (bytes, 256-aligned regions)
  char* ws = (char*)d_ws;
  __bf16*   Win_b  = (__bf16*)  (ws + 0);          // 1024*256*2  = 512 KB
  __bf16*   Wih_b  = (__bf16*)  (ws + 524288);     // 1024*1024*2 =   2 MB
  __bf16*   Whh_b  = (__bf16*)  (ws + 2621440);    //               2 MB
  __bf16*   Wout_b = (__bf16*)  (ws + 4718592);    //             512 KB
  float*    bias2  = (float*)   (ws + 5242880);    // 4 KB (b_ih+b_hh)
  unsigned* ctr    = (unsigned*)(ws + 5246976);    // barrier counter
  __bf16*   Hbuf   = (__bf16*)  (ws + 5247232);    // 2 x [B,NHID] bf16 ping-pong
  __bf16*   X1     = (__bf16*)  (ws + 5509376);    // [T*B,NHID] bf16 = 64 MB

  const int MROWS = T_STEPS * BATCH;  // 32768

  // Phase 0: weight conversion (recomputed every call -> deterministic)
  cvt_f32_bf16<<<(NHID * NIN  + 255) / 256, 256, 0, stream>>>(W_in,  Win_b,  NHID * NIN);
  cvt_f32_bf16<<<(NHID * NHID + 255) / 256, 256, 0, stream>>>(W_ih,  Wih_b,  NHID * NHID);
  cvt_f32_bf16<<<(NHID * NHID + 255) / 256, 256, 0, stream>>>(W_hh,  Whh_b,  NHID * NHID);
  cvt_f32_bf16<<<(NOUT * NHID + 255) / 256, 256, 0, stream>>>(W_out, Wout_b, NOUT * NHID);
  prep_bias_h0<<<(BATCH * NHID + 255) / 256, 256, 0, stream>>>(b_ih, b_hh, bias2, Hbuf, ctr);

  // Phase 1: X1 = relu(X @ W_in^T + b_in), bf16   (A is f32, K=256, N=1024)
  gemm_bf16_wmma<NIN, NHID, 4, true, true, true, false, true>
      <<<dim3(NHID / 256, MROWS / 16), 128, 0, stream>>>(X, Win_b, b_in, nullptr, X1);

  // Phase 2: U = X1 @ W_ih^T + (b_ih + b_hh) -> d_out hiddens (f32)
  gemm_bf16_wmma<NHID, NHID, 4, false, false, true, true, false>
      <<<dim3(NHID / 256, MROWS / 16), 128, 0, stream>>>(X1, Wih_b, bias2, hiddens, nullptr);

  // Phase 3: persistent scan, single launch, grid barrier per time step
  rnn_scan_persistent<<<dim3(SCAN_WGS), 128, 0, stream>>>(Whh_b, hiddens, Hbuf, ctr);

  // Phase 4: O = H @ W_out^T + b_out (A = f32 hiddens from d_out, N=256)
  gemm_bf16_wmma<NHID, NOUT, 4, true, false, true, true, false>
      <<<dim3(NOUT / 256, MROWS / 16), 128, 0, stream>>>(hiddens, Wout_b, b_out, outputs, nullptr);
}
